// InteractionAttention_68728066670695
// MI455X (gfx1250) — matrix operand order repaired and checked by reading, not compiled, not run on hardware
//
#include <hip/hip_runtime.h>
#include <hip/hip_bf16.h>
#include <stdint.h>

// ---------------------------------------------------------------------------
// InteractionAttention for MI455X (gfx1250, wave32, WMMA bf16 16x16x32).
// All matrix math through v_wmma_f32_16x16x32_bf16, f32 accumulation.
// Channel-major intermediates [c, b, spatial]; fp32->bf16 conversion fused
// into LDS staging. Fully templated strides/dtypes so staging is straight-line
// vectorized loads (no divergent per-element branches, no serialized waits),
// with double-buffered LDS to overlap global traffic with WMMA.
// Workspace requirement: ~336 MB.
// ---------------------------------------------------------------------------

typedef __attribute__((ext_vector_type(16))) __bf16        v16bf;
typedef __attribute__((ext_vector_type(8)))  float         v8f;
typedef __attribute__((ext_vector_type(4)))  unsigned int  u32x4;
typedef __attribute__((ext_vector_type(4)))  float         f32x4;
typedef __attribute__((ext_vector_type(4)))  unsigned short u16x4;

union FragAB { v16bf v; u32x4 q[2]; unsigned short s[16]; };

__device__ __forceinline__ unsigned short f2bf_bits(float f) {
    union { float f; unsigned u; } a; a.f = f;
    unsigned u = a.u;
    u += 0x7FFFu + ((u >> 16) & 1u);            // round-to-nearest-even
    return (unsigned short)(u >> 16);
}
__device__ __forceinline__ __bf16 f2bf(float f) {
    union { unsigned short s; __bf16 b; } o; o.s = f2bf_bits(f); return o.b;
}
__device__ __forceinline__ float bf2f(unsigned short s) {
    union { unsigned u; float f; } o; o.u = (unsigned)s << 16; return o.f;
}

// ---------------------------------------------------------------------------
// Big tiled GEMM:  C[m,n] = sum_k A[m,k]*B[k,n] + bias[m]
// BM=BN=128, BK=32; 256 threads = 8 waves (4x2), wave tile 32x64 (2x4 frags).
// Template: ABF/BBF/CBF dtype flags (bf16=1), compile-time strides & shapes.
// Requires SAM==1 (true for every call: A is a weight matrix read along m).
// ---------------------------------------------------------------------------
#define BM 128
#define BN 128
#define BKK 32
#define LDAP 40   /* BK+8 bf16 -> 80B rows: 16B aligned, bank-spread */
#define LDBP 40

template<int ABF, int BBF, int CBF,
         long long SAM, long long SAK,
         long long SBK, long long SBN,
         long long SCM, long long SCN,
         int MM, int NN, int KK>
__global__ void __launch_bounds__(256)
gemm_tiled(const void* __restrict__ Ap, const void* __restrict__ Bp,
           const float* __restrict__ bias, void* __restrict__ Cp)
{
    static_assert(ABF == 0 && SAM == 1, "A operand: fp32 weights, unit m-stride");
    static_assert(KK % BKK == 0 && MM % BM == 0 && NN % BN == 0, "exact tiling");

    __shared__ __bf16 la[2][BM * LDAP];
    __shared__ __bf16 lb[2][BN * LDBP];

    const int tid  = threadIdx.x;
    const int lane = tid & 31;
    const int w    = tid >> 5;
    const int wm   = w >> 1;        // 0..3
    const int wn   = w & 1;         // 0..1
    const int half = lane >> 4;
    const int lr   = lane & 15;

    const long long m0 = (long long)blockIdx.y * BM;
    const long long n0 = (long long)blockIdx.x * BN;

    const float* __restrict__ Af = (const float*)Ap;
    const float* __restrict__ Bf = (const float*)Bp;
    const unsigned short* __restrict__ Bh = (const unsigned short*)Bp;

    v8f acc[2][4];
    const v8f vzero = {0.f,0.f,0.f,0.f,0.f,0.f,0.f,0.f};
    #pragma unroll
    for (int i = 0; i < 2; ++i)
        #pragma unroll
        for (int j = 0; j < 4; ++j) acc[i][j] = vzero;

    f32x4 aReg[4];          // 16 fp32 A elements per thread
    f32x4 bRegF[4];         // 16 B elements per thread (fp32 source)
    u16x4 bRegH[4];         // (bf16 source)

    // ---- staging: registers <- global ----
    auto loadA = [&](int kb) {
        #pragma unroll
        for (int i = 0; i < 4; ++i) {
            int item = tid + i * 256;            // 0..1023
            int m4 = (item & 31) << 2;           // 0,4,...,124
            int k  = item >> 5;                  // 0..31
            aReg[i] = *(const f32x4*)&Af[(m0 + m4) + (long long)(kb + k) * SAK];
        }
    };
    auto loadB = [&](int kb) {
        if constexpr (!BBF && SBK == 1) {        // k contiguous (GEMM1: B = x)
            #pragma unroll
            for (int i = 0; i < 4; ++i) {
                int item = tid + i * 256;
                int n  = item & 127;
                int kg = item >> 7;              // 0..7 (groups of 4 k)
                bRegF[i] = *(const f32x4*)&Bf[(n0 + n) * SBN + kb + kg * 4];
            }
        } else if constexpr (!BBF && SBN == 1) { // n contiguous, fp32
            #pragma unroll
            for (int i = 0; i < 4; ++i) {
                int item = tid + i * 256;
                int n4 = (item & 31) << 2;
                int k  = item >> 5;
                bRegF[i] = *(const f32x4*)&Bf[(long long)(kb + k) * SBK + n0 + n4];
            }
        } else {                                  // n contiguous, bf16 (V path)
            static_assert(BBF && SBN == 1, "unsupported B layout");
            #pragma unroll
            for (int i = 0; i < 4; ++i) {
                int item = tid + i * 256;
                int n4 = (item & 31) << 2;
                int k  = item >> 5;
                bRegH[i] = *(const u16x4*)&Bh[(long long)(kb + k) * SBK + n0 + n4];
            }
        }
    };
    // ---- staging: LDS <- registers ----
    auto storeA = [&](int buf) {
        #pragma unroll
        for (int i = 0; i < 4; ++i) {
            int item = tid + i * 256;
            int m4 = (item & 31) << 2;
            int k  = item >> 5;
            #pragma unroll
            for (int j = 0; j < 4; ++j)
                la[buf][(m4 + j) * LDAP + k] = f2bf(aReg[i][j]);
        }
    };
    auto storeB = [&](int buf) {
        if constexpr (!BBF && SBK == 1) {
            #pragma unroll
            for (int i = 0; i < 4; ++i) {
                int item = tid + i * 256;
                int n  = item & 127;
                int kg = item >> 7;
                u16x4 h;
                #pragma unroll
                for (int j = 0; j < 4; ++j) h[j] = f2bf_bits(bRegF[i][j]);
                *(u16x4*)&lb[buf][n * LDBP + kg * 4] = h;   // 8B ds_store
            }
        } else if constexpr (!BBF && SBN == 1) {
            #pragma unroll
            for (int i = 0; i < 4; ++i) {
                int item = tid + i * 256;
                int n4 = (item & 31) << 2;
                int k  = item >> 5;
                #pragma unroll
                for (int j = 0; j < 4; ++j)
                    lb[buf][(n4 + j) * LDBP + k] = f2bf(bRegF[i][j]);
            }
        } else {
            #pragma unroll
            for (int i = 0; i < 4; ++i) {
                int item = tid + i * 256;
                int n4 = (item & 31) << 2;
                int k  = item >> 5;
                #pragma unroll
                for (int j = 0; j < 4; ++j)
                    ((unsigned short*)lb[buf])[(n4 + j) * LDBP + k] = bRegH[i][j];
            }
        }
    };

    constexpr int ksteps = KK / BKK;
    loadA(0); loadB(0);
    storeA(0); storeB(0);
    __syncthreads();

    int buf = 0;
    #pragma unroll 1
    for (int kk = 0; kk < ksteps; ++kk) {
        if (kk + 1 < ksteps) {                   // issue next tile's global loads
            loadA((kk + 1) * BKK);
            loadB((kk + 1) * BKK);
        }
        FragAB af[2], bf[4];
        #pragma unroll
        for (int i = 0; i < 2; ++i) {
            int m = wm * 32 + i * 16 + lr;
            af[i].q[0] = *(const u32x4*)&la[buf][m * LDAP + half * 8];
            af[i].q[1] = *(const u32x4*)&la[buf][m * LDAP + 16 + half * 8];
        }
        #pragma unroll
        for (int j = 0; j < 4; ++j) {
            int n = wn * 64 + j * 16 + lr;
            bf[j].q[0] = *(const u32x4*)&lb[buf][n * LDBP + half * 8];
            bf[j].q[1] = *(const u32x4*)&lb[buf][n * LDBP + 16 + half * 8];
        }
        #pragma unroll
        for (int i = 0; i < 2; ++i)
            #pragma unroll
            for (int j = 0; j < 4; ++j)
                acc[i][j] = __builtin_amdgcn_wmma_f32_16x16x32_bf16(
                    false, af[i].v, false, bf[j].v, (short)0, acc[i][j], false, false);
        if (kk + 1 < ksteps) {
            storeA(buf ^ 1);
            storeB(buf ^ 1);
        }
        __syncthreads();
        buf ^= 1;
    }

    // epilogue: C VGPR r -> M = r + 8*half, N = lane&15
    #pragma unroll
    for (int i = 0; i < 2; ++i)
        #pragma unroll
        for (int j = 0; j < 4; ++j) {
            long long mBase = m0 + wm * 32 + i * 16 + half * 8;
            long long n     = n0 + wn * 64 + j * 16 + lr;
            #pragma unroll
            for (int r = 0; r < 8; ++r) {
                long long m = mBase + r;
                float v = acc[i][j][r] + bias[m];
                if constexpr (CBF) ((unsigned short*)Cp)[m * SCM + n * SCN] = f2bf_bits(v);
                else               ((float*)Cp)[m * SCM + n * SCN] = v;
            }
        }
}

// ---------------------------------------------------------------------------
// Batched small GEMM: per batch g, C = A(MxK) @ B(KxN), fp32 out.
// 4 waves per workgroup; 16x16 output tiles round-robin per wave.
// bf16 unit-k-stride operands load fragments as two global b128s (fp32 unit-k A: four, converted); otherwise
// scalar gathers with K-guard (zero pad) — handles the K=16 final einsum.
// Output base = SCC*(g/8) + SCB*(g%8) (scatter into [b,c,...] for d_out).
// ---------------------------------------------------------------------------
template<int ABF, int BBF,
         long long SAG, long long SAM, long long SAK,
         long long SBG, long long SBK, long long SBN,
         long long SCC, long long SCB, long long SCM, long long SCN,
         int M, int N, int K>
__global__ void __launch_bounds__(128)
gemm_small(const void* __restrict__ Ap, const void* __restrict__ Bp,
           float* __restrict__ Cp)
{
    const int g    = blockIdx.x;
    const int tid  = threadIdx.x;
    const int lane = tid & 31;
    const int w    = tid >> 5;
    const int half = lane >> 4;
    const int lr   = lane & 15;

    constexpr int mtN = M >> 4, ntN = N >> 4, tt = mtN * ntN;
    constexpr int ksteps = (K + 31) >> 5;
    const long long abase = (long long)g * SAG;
    const long long bbase = (long long)g * SBG;
    const long long cbase = SCC * (long long)(g >> 3) + SCB * (long long)(g & 7);
    const v8f vzero = {0.f,0.f,0.f,0.f,0.f,0.f,0.f,0.f};

    const float* __restrict__ Af = (const float*)Ap;
    const float* __restrict__ Bf = (const float*)Bp;
    const unsigned short* __restrict__ Ah = (const unsigned short*)Ap;
    const unsigned short* __restrict__ Bh = (const unsigned short*)Bp;

    #pragma unroll
    for (int t = w; t < tt; t += 4) {
        const int mt = t / ntN;
        const int nt = t % ntN;
        v8f acc = vzero;
        #pragma unroll
        for (int kk = 0; kk < ksteps; ++kk) {
            const int kb = kk * 32;
            FragAB a, b;
            const int m = mt * 16 + lr;
            if constexpr (ABF && SAK == 1 && (K % 32) == 0) {
                const unsigned short* p = Ah + abase + (long long)m * SAM + kb;
                a.q[0] = *(const u32x4*)(p + half * 8);
                a.q[1] = *(const u32x4*)(p + 16 + half * 8);
            } else if constexpr (!ABF && SAK == 1 && (K % 32) == 0) {
                const float* p = Af + abase + (long long)m * SAM + kb;
                f32x4 lo0 = *(const f32x4*)(p + half * 8);
                f32x4 lo1 = *(const f32x4*)(p + half * 8 + 4);
                f32x4 hi0 = *(const f32x4*)(p + 16 + half * 8);
                f32x4 hi1 = *(const f32x4*)(p + 16 + half * 8 + 4);
                #pragma unroll
                for (int j = 0; j < 4; ++j) {
                    a.s[j]      = f2bf_bits(lo0[j]);
                    a.s[4 + j]  = f2bf_bits(lo1[j]);
                    a.s[8 + j]  = f2bf_bits(hi0[j]);
                    a.s[12 + j] = f2bf_bits(hi1[j]);
                }
            } else {
                #pragma unroll
                for (int j = 0; j < 8; ++j) {
                    int k0 = kb + ((j < 4) ? 0 : 16) + half * 8 + (j & 3) * 2;
                    float f0 = 0.f, f1 = 0.f;
                    if (k0 < K)
                        f0 = ABF ? bf2f(Ah[abase + (long long)m * SAM + (long long)k0 * SAK])
                                 : Af[abase + (long long)m * SAM + (long long)k0 * SAK];
                    if (k0 + 1 < K)
                        f1 = ABF ? bf2f(Ah[abase + (long long)m * SAM + (long long)(k0 + 1) * SAK])
                                 : Af[abase + (long long)m * SAM + (long long)(k0 + 1) * SAK];
                    a.s[2 * j]     = f2bf_bits(f0);
                    a.s[2 * j + 1] = f2bf_bits(f1);
                }
            }
            const int n = nt * 16 + lr;
            if constexpr (BBF && SBK == 1 && (K % 32) == 0) {
                const unsigned short* p = Bh + bbase + (long long)n * SBN + kb;
                b.q[0] = *(const u32x4*)(p + half * 8);
                b.q[1] = *(const u32x4*)(p + 16 + half * 8);
            } else {
                #pragma unroll
                for (int e = 0; e < 16; ++e) {
                    int k = kb + ((e < 8) ? 0 : 16) + half * 8 + (e & 7);
                    float f = 0.f;
                    if (k < K)
                        f = BBF ? bf2f(Bh[bbase + (long long)k * SBK + (long long)n * SBN])
                                : Bf[bbase + (long long)k * SBK + (long long)n * SBN];
                    b.s[e] = f2bf_bits(f);
                }
            }
            acc = __builtin_amdgcn_wmma_f32_16x16x32_bf16(
                false, a.v, false, b.v, (short)0, acc, false, false);
        }
        #pragma unroll
        for (int r = 0; r < 8; ++r) {
            long long m = mt * 16 + half * 8 + r;
            long long n = nt * 16 + lr;
            Cp[cbase + m * SCM + n * SCN] = acc[r];
        }
    }
}

// ---------------------------------------------------------------------------
// avg-pool N by PS=4 on bf16 channel-major Q/K (channels [0,512), [512,1024))
// ---------------------------------------------------------------------------
__global__ void __launch_bounds__(256)
pool_kernel(const unsigned short* __restrict__ qkv,
            unsigned short* __restrict__ qp, unsigned short* __restrict__ kp)
{
    long long t = (long long)blockIdx.x * 256 + threadIdx.x;   // 512*8*16*64 total
    int d = (int)(t & 63);
    int l = (int)((t >> 6) & 15);
    int b = (int)((t >> 10) & 7);
    long long c = t >> 13;
    long long base = c * 32768 + (long long)b * 4096 + (long long)l * 4 * 64 + d;
    float sq = 0.f, sk = 0.f;
    #pragma unroll
    for (int p = 0; p < 4; ++p) {
        sq += bf2f(qkv[base + p * 64]);
        sk += bf2f(qkv[512LL * 32768 + base + p * 64]);
    }
    long long og = (c * 8 + b) * 1024 + l * 64 + d;
    qp[og] = f2bf_bits(sq * 0.25f);
    kp[og] = f2bf_bits(sk * 0.25f);
}

// ---------------------------------------------------------------------------
// In-place softmax over 512 channels, channel-major [512][ncols]; one thread
// per column (coalesced across threads within each channel row).
// ---------------------------------------------------------------------------
__global__ void __launch_bounds__(256)
softmax_ch(float* __restrict__ data, int ncols)
{
    long long col = (long long)blockIdx.x * 256 + threadIdx.x;
    if (col >= ncols) return;
    float m = -3.4e38f, s = 0.f;
    for (int c = 0; c < 512; ++c) {
        float v = data[(long long)c * ncols + col];
        float nm = fmaxf(m, v);
        s = s * __expf(m - nm) + __expf(v - nm);
        m = nm;
    }
    float inv = 1.0f / s;
    for (int c = 0; c < 512; ++c) {
        long long i = (long long)c * ncols + col;
        data[i] = __expf(data[i] - m) * inv;
    }
}

// ---------------------------------------------------------------------------
extern "C" void kernel_launch(void* const* d_in, const int* in_sizes, int n_in,
                              void* d_out, int out_size, void* d_ws, size_t ws_size,
                              hipStream_t stream)
{
    (void)in_sizes; (void)n_in; (void)out_size; (void)ws_size;

    const float* x       = (const float*)d_in[0];   // [32768, 512]
    const float* W_in    = (const float*)d_in[1];   // [512, 1536]
    const float* b_in    = (const float*)d_in[2];   // [1536]
    const float* W_inter = (const float*)d_in[3];   // [512, 512]
    const float* b_inter = (const float*)d_in[4];   // [512]
    const float* W_last  = (const float*)d_in[5];   // [512, 512]
    const float* b_last  = (const float*)d_in[6];   // [512]
    float* out = (float*)d_out;                     // [8, 512, 64, 64]

    char* ws = (char*)d_ws;
    size_t o = 0;
    unsigned short* qkv = (unsigned short*)(ws + o); o += 1536LL * 32768 * 2;  // bf16 ch-major
    unsigned short* qp  = (unsigned short*)(ws + o); o += 4194304LL * 2;
    unsigned short* kp  = (unsigned short*)(ws + o); o += 4194304LL * 2;
    float* qk_raw       = (float*)(ws + o); o += 4194304LL * 4;  // reused later as qKV
    float* qk_int       = (float*)(ws + o); o += 4194304LL * 4;
    float* qk_out       = (float*)(ws + o); o += 4194304LL * 4;
    float* qK_raw       = (float*)(ws + o); o += 4194304LL * 4;
    float* qK_int       = (float*)(ws + o); o += 4194304LL * 4;
    float* qK_out       = (float*)(ws + o); o += 4194304LL * 4;
    float* vp_int       = (float*)(ws + o); o += 16777216LL * 4;
    float* vp_out       = (float*)(ws + o); o += 16777216LL * 4;
    float* qkv2 = qk_raw;  // qKV reuses qk_raw (dead after qk_int)

    // 1) GEMM1: qkv[k, col] = sum_c W_in[c,k] * x[col,c] + b_in[k]   (bf16 out)
    gemm_tiled<0,0,1, 1,1536, 1,512, 32768,1, 1536,32768,512>
        <<<dim3(32768/128, 1536/128), 256, 0, stream>>>(W_in, x, b_in, qkv);

    // 2) pool Q,K over N (PS=4)
    pool_kernel<<<16384, 256, 0, stream>>>(qkv, qp, kp);

    // 3) Qk[b,c,n,l] = sum_d Q[n,d] * k_pool[l,d]
    gemm_small<1,1, 4096,64,1, 1024,1,64, 8192,1024,16,1, 64,16,64>
        <<<4096, 128, 0, stream>>>(qkv, kp, qk_raw);

    // 4) inter+softmax+last on Qk  (ncols = 8192)
    gemm_tiled<0,0,0, 1,512, 8192,1, 8192,1, 512,8192,512>
        <<<dim3(8192/128, 512/128), 256, 0, stream>>>(W_inter, qk_raw, b_inter, qk_int);
    softmax_ch<<<32, 256, 0, stream>>>(qk_int, 8192);
    gemm_tiled<0,0,0, 1,512, 8192,1, 8192,1, 512,8192,512>
        <<<dim3(8192/128, 512/128), 256, 0, stream>>>(W_last, qk_int, b_last, qk_out);

    // 5) qK[b,c,l,n] = sum_d q_pool[l,d] * K[n,d]; then inter+softmax+last
    gemm_small<1,1, 1024,64,1, 4096,1,64, 8192,1024,64,1, 16,64,64>
        <<<4096, 128, 0, stream>>>(qp, qkv + 512LL * 32768, qK_raw);
    gemm_tiled<0,0,0, 1,512, 8192,1, 8192,1, 512,8192,512>
        <<<dim3(8192/128, 512/128), 256, 0, stream>>>(W_inter, qK_raw, b_inter, qK_int);
    softmax_ch<<<32, 256, 0, stream>>>(qK_int, 8192);
    gemm_tiled<0,0,0, 1,512, 8192,1, 8192,1, 512,8192,512>
        <<<dim3(8192/128, 512/128), 256, 0, stream>>>(W_last, qK_int, b_last, qK_out);

    // 6) Vp = last(softmax(inter(V)))   (ncols = 32768, V = channels [1024,1536))
    gemm_tiled<0,1,0, 1,512, 32768,1, 32768,1, 512,32768,512>
        <<<dim3(32768/128, 512/128), 256, 0, stream>>>(W_inter, qkv + 1024LL * 32768, b_inter, vp_int);
    softmax_ch<<<128, 256, 0, stream>>>(vp_int, 32768);
    gemm_tiled<0,0,0, 1,512, 32768,1, 32768,1, 512,32768,512>
        <<<dim3(32768/128, 512/128), 256, 0, stream>>>(W_last, vp_int, b_last, vp_out);

    // 7) qKV[b,c,l,d] = sum_n qK_out[l,n] * Vp[n,d]
    gemm_small<0,0, 1024,64,1, 4096,64,1, 8192,1024,64,1, 16,64,64>
        <<<4096, 128, 0, stream>>>(qK_out, vp_out, qkv2);

    // 8) out[b,c,n,d] = sum_l Qk_out[n,l] * qKV[l,d]   (K=16, zero-padded)
    gemm_small<0,0, 1024,16,1, 1024,64,1, 4096,2097152,64,1, 64,64,16>
        <<<4096, 128, 0, stream>>>(qk_out, qkv2, out);
}
